// GatedDeltaNetLayer_65103114272832
// MI455X (gfx1250) — matrix operand order repaired and checked by reading, not compiled, not run on hardware
//
#include <hip/hip_runtime.h>
#include <hip/hip_bf16.h>
#include <stdint.h>

// Problem dims (fixed by reference)
#define B_     4
#define T_     2048
#define DM_    1024
#define H_     16
#define D_     64
#define INNER_ 1024
#define ROWS_  (B_ * T_)   // 8192

typedef __attribute__((ext_vector_type(16))) __bf16 v16bf;
typedef __attribute__((ext_vector_type(8)))  float  v8f;
typedef __attribute__((ext_vector_type(4)))  unsigned int u32x4;
typedef __attribute__((ext_vector_type(8)))  int  i32x8;
typedef __attribute__((ext_vector_type(4)))  int  i32x4;

#if __has_builtin(__builtin_amdgcn_tensor_load_to_lds) && \
    __has_builtin(__builtin_amdgcn_s_wait_tensorcnt)
#define HAS_TDM 1
#endif

__device__ __forceinline__ __bf16 f2bf(float f) {
  unsigned u = __builtin_bit_cast(unsigned, f);
  u += 0x7fffu + ((u >> 16) & 1u);                 // round-to-nearest-even
  unsigned short h = (unsigned short)(u >> 16);
  return __builtin_bit_cast(__bf16, h);
}
__device__ __forceinline__ float bf2f(__bf16 b) {
  unsigned short h = __builtin_bit_cast(unsigned short, b);
  unsigned u = ((unsigned)h) << 16;
  return __builtin_bit_cast(float, u);
}

union FragU { uint4 q[2]; v16bf v; };

// ---- WMMA fragment loaders (wave32), contiguous k order k = 16h + s, same for A and B ----
// A (16x32 bf16): lane row m; lanes<16: K 0..15, lanes>=16: K 16..31
__device__ __forceinline__ v16bf a_frag_bf(const __bf16* row, int ks, int lane) {
  const __bf16* p = row + ks + ((lane & 16) ? 16 : 0);
  FragU u;
  u.q[0] = *(const uint4*)(p);
  u.q[1] = *(const uint4*)(p + 8);
  return u.v;
}
// B (32x16 bf16): lane col n = lane&15; lanes<16: K 0..15, lanes>=16: K 16..31.
// "contig" variant: B supplied as B-transpose rows (column n contiguous in K).
__device__ __forceinline__ v16bf b_frag_bf(const __bf16* row, int ks, int lane) {
  const __bf16* p = row + ks + ((lane & 16) ? 16 : 0);
  FragU u;
  u.q[0] = *(const uint4*)(p);
  u.q[1] = *(const uint4*)(p + 8);
  return u.v;
}
__device__ __forceinline__ v16bf b_frag_f32(const float* row, int ks, int lane) {
  const float* p = row + ks + ((lane & 16) ? 16 : 0);
  v16bf v;
#pragma unroll
  for (int i = 0; i < 16; ++i) v[i] = f2bf(p[i]);
  return v;
}
__device__ __forceinline__ v16bf b_frag_strided(const __bf16* colp, int stride, int ks, int lane) {
  const __bf16* p = colp + (size_t)(ks + ((lane & 16) ? 16 : 0)) * stride;
  v16bf v;
#pragma unroll
  for (int e = 0; e < 16; ++e) v[e] = p[(size_t)e * stride];
  return v;
}

__device__ __forceinline__ v8f wmma_bf16(v16bf a, v16bf b, v8f c) {
  return __builtin_amdgcn_wmma_f32_16x16x32_bf16(false, a, false, b, (short)0, c, false, false);
}

#ifdef HAS_TDM
// TDM: DMA a 64x64 bf16 tile (row stride 1024 elements) from global into LDS.
// D# packing per CDNA5 ISA §8.3 (group0) / §8.4 (group1).
// This toolchain exposes the 6-arg builtin form (extra i32x8 group + cpol).
__device__ __forceinline__ void tdm_load_tile_bf16(const __bf16* gsrc, void* lds_dst) {
  unsigned long long ga = (unsigned long long)gsrc;
  unsigned lds = (unsigned)(unsigned long long)lds_dst;   // low 32 bits = LDS offset
  u32x4 g0 = { 1u,                                   // count=1, user mode
               lds,                                  // lds_addr
               (unsigned)ga,                         // global_addr[31:0]
               (unsigned)((ga >> 32) & 0x1ffffffu) | 0x80000000u }; // addr[56:32] | type=2
  i32x8 g1 = { (int)(1u << 16),       // workgroup_mask=0, data_size=1 (2B)
               (int)(64u << 16),      // atomic_bar=0 | tensor_dim0 lo16 = 64
               (int)(64u << 16),      // tensor_dim0 hi=0 | tensor_dim1 lo16 = 64
               (int)(64u << 16),      // tensor_dim1 hi=0 | tile_dim0 = 64
               (int)64,               // tile_dim1 = 64, tile_dim2 = 0
               (int)1024,             // tensor_dim0_stride lo32 = 1024 elements
               0, 0 };                // stride0 hi / stride1 (unused for 2D)
  __builtin_amdgcn_tensor_load_to_lds(g0, g1,
                                      (i32x4){0, 0, 0, 0},
                                      (i32x4){0, 0, 0, 0},
                                      (i32x8){0, 0, 0, 0, 0, 0, 0, 0},
                                      0);
}
#endif

// ---------------- fp32 -> bf16 convert ----------------
__global__ __launch_bounds__(256)
void f32_to_bf16_kernel(const float* __restrict__ in, __bf16* __restrict__ out, size_t n) {
  size_t i = (size_t)blockIdx.x * blockDim.x + threadIdx.x;
  size_t stride = (size_t)gridDim.x * blockDim.x;
  for (; i < n; i += stride) out[i] = f2bf(in[i]);
}

// ---------------- gate projections: alpha/beta = sigmoid(x@W + b) ----------------
__global__ __launch_bounds__(256)
void gates_kernel(const float* __restrict__ x,
                  const float* __restrict__ Wa, const float* __restrict__ ba,
                  const float* __restrict__ Wb, const float* __restrict__ bb,
                  float* __restrict__ alpha, float* __restrict__ beta) {
  int id = blockIdx.x * blockDim.x + threadIdx.x;
  if (id >= ROWS_ * H_) return;
  int row = id / H_, h = id % H_;
  const float* xr = x + (size_t)row * DM_;
  float sa = ba[h], sb = bb[h];
#pragma unroll 4
  for (int d = 0; d < DM_; ++d) {
    float xv = xr[d];
    sa += xv * Wa[(size_t)d * H_ + h];
    sb += xv * Wb[(size_t)d * H_ + h];
  }
  alpha[id] = 1.0f / (1.0f + __expf(-sa));
  beta[id]  = 1.0f / (1.0f + __expf(-sb));
}

// ---------------- big bf16 WMMA GEMM: C[M,N] = A[M,K] @ B[K,N] ----------------
// block: 256 thr = 8 waves; tile 128x64; wave tile 16x64 (4 accumulators)
template<bool OUT_BF16>
__global__ __launch_bounds__(256)
void gemm_bf16_kernel(const __bf16* __restrict__ A, const __bf16* __restrict__ Bm,
                      void* __restrict__ Cv, int M, int N, int K) {
  __shared__ __bf16 btile[4 * 32 * 16];   // B staged directly in fragment layout
  const int tid  = threadIdx.x;
  const int wave = tid >> 5, lane = tid & 31;
  const int colBase = blockIdx.x * 64;
  const int rowBase = blockIdx.y * 128 + wave * 16;

  v8f acc[4];
#pragma unroll
  for (int nt = 0; nt < 4; ++nt) acc[nt] = (v8f){0, 0, 0, 0, 0, 0, 0, 0};

  const int kl   = tid >> 3;             // 0..31 (k-row in tile)
  const int n0   = (tid & 7) * 8;        // 0..56 (n start, 8 elems)
  const int e    = kl & 15;
  const int lsel = (kl & 16) ? 16 : 0;

  const __bf16* arow = A + (size_t)(rowBase + (lane & 15)) * K;

  for (int k = 0; k < K; k += 32) {
    // stage 32x64 B tile into fragment layout
    uint4 braw = *(const uint4*)(Bm + (size_t)(k + kl) * N + colBase + n0);
    union { uint4 q; unsigned short s[8]; } bu; bu.q = braw;
#pragma unroll
    for (int j = 0; j < 8; ++j) {
      int nl = n0 + j;
      btile[(((nl >> 4) * 32) + (nl & 15) + lsel) * 16 + e] =
          __builtin_bit_cast(__bf16, bu.s[j]);
    }
    __syncthreads();
    __builtin_prefetch(arow + k + 64, 0, 1);        // global_prefetch next A slab
    v16bf af = a_frag_bf(arow, k, lane);
#pragma unroll
    for (int nt = 0; nt < 4; ++nt) {
      const __bf16* bp = &btile[(nt * 32 + lane) * 16];
      FragU u;
      u.q[0] = *(const uint4*)(bp);
      u.q[1] = *(const uint4*)(bp + 8);
      acc[nt] = wmma_bf16(af, u.v, acc[nt]);
    }
    __syncthreads();
  }
#pragma unroll
  for (int nt = 0; nt < 4; ++nt) {
#pragma unroll
    for (int r = 0; r < 8; ++r) {
      int row = rowBase + r + ((lane & 16) ? 8 : 0);
      int col = colBase + nt * 16 + (lane & 15);
      if (OUT_BF16) ((__bf16*)Cv)[(size_t)row * N + col] = f2bf(acc[nt][r]);
      else          ((float*)Cv)[(size_t)row * N + col]  = acc[nt][r];
    }
  }
}

// ---------------- chunked gated-linear-attention scan ----------------
// one block per (b,h); 128 thr = 4 waves; chunk L=64; TDM chunk staging;
// all 4 per-chunk GEMMs on WMMA
__global__ __launch_bounds__(128)
void scan_kernel(const __bf16* __restrict__ Qb, const __bf16* __restrict__ Kb,
                 const __bf16* __restrict__ Vb, const float* __restrict__ alpha,
                 const float* __restrict__ beta, const float* __restrict__ S0,
                 __bf16* __restrict__ Yb, float* __restrict__ Sout) {
  __shared__ __bf16 Qs[64][64];    // TDM raw Q -> scaled in place: Q * A_t
  __shared__ __bf16 Kt[64][64];    // TDM raw K -> scaled in place: K * beta/A_s
  __shared__ __bf16 Vraw[64][64];  // TDM raw V (row major)
  __shared__ __bf16 Vt[64][64];    // V transposed
  __shared__ __bf16 Pm[64][64];    // masked attention (bf16: WMMA operand only)
  __shared__ float  Sm[64][64];    // running state (fp32)
  __shared__ float  la[64], kscale[64], bw[64];

  const int bh = blockIdx.x;
  const int b = bh / H_, h = bh % H_;
  const int tid = threadIdx.x;
  const int wave = tid >> 5, lane = tid & 31;
  const int m0 = wave * 16;     // each wave owns a 16-row strip

  for (int idx = tid; idx < 4096; idx += 128)
    Sm[idx >> 6][idx & 63] =
        S0[(size_t)((b * H_ + h) * 64 + (idx >> 6)) * 64 + (idx & 63)];
  __syncthreads();

  for (int chunk = 0; chunk < T_ / 64; ++chunk) {
    const int t0 = chunk * 64;
    const size_t base = (size_t)(b * T_ + t0) * INNER_ + h * 64;

    // --- stage raw Q/K/V chunk into LDS ---
#ifdef HAS_TDM
    if (wave == 0) {                       // one wave issues the 3 DMA descriptors
      tdm_load_tile_bf16(Qb + base, &Qs[0][0]);
      tdm_load_tile_bf16(Kb + base, &Kt[0][0]);
      tdm_load_tile_bf16(Vb + base, &Vraw[0][0]);
      __builtin_amdgcn_s_wait_tensorcnt(0);
    }
    __syncthreads();
#else
    for (int idx = tid; idx < 4096; idx += 128) {
      int r = idx >> 6, cc = idx & 63;
      size_t g = base + (size_t)r * INNER_ + cc;
      Qs[r][cc]   = Qb[g];
      Kt[r][cc]   = Kb[g];
      Vraw[r][cc] = Vb[g];
    }
    __syncthreads();
#endif

    // --- per-row scalars: log-decay, write gate, K row norm (from raw LDS K) ---
    if (tid < 64) {
      size_t gidx = (size_t)(b * T_ + t0 + tid) * H_ + h;
      la[tid] = __logf(fmaxf(alpha[gidx], 1e-30f));
      bw[tid] = beta[gidx];
      float ss = 0.f;
#pragma unroll 4
      for (int j = 0; j < 64; ++j) { float v = bf2f(Kt[tid][j]); ss += v * v; }
      kscale[tid] = 1.0f / fmaxf(sqrtf(ss), 1e-12f);
    }
    __syncthreads();
    if (tid == 0) {                       // inclusive prefix of log-decay
      float c = 0.f;
      for (int r = 0; r < 64; ++r) { c += la[r]; la[r] = c; }
    }
    __syncthreads();
    // --- scale Q,K in place; build transposed V ---
    for (int idx = tid; idx < 4096; idx += 128) {
      int r = idx >> 6, cc = idx & 63;
      Qs[r][cc] = f2bf(bf2f(Qs[r][cc]) * __expf(la[r]));
      Kt[r][cc] = f2bf(bf2f(Kt[r][cc]) * kscale[r] * bw[r] * __expf(-la[r]));
      Vt[cc][r] = Vraw[r][cc];
    }
    __syncthreads();

    // --- GEMM1: P = Qs @ Kt^T, causal mask on bf16 store ---
#pragma unroll
    for (int nt = 0; nt < 4; ++nt) {
      v8f acc = (v8f){0, 0, 0, 0, 0, 0, 0, 0};
#pragma unroll
      for (int ks = 0; ks < 64; ks += 32)
        acc = wmma_bf16(a_frag_bf(&Qs[m0 + (lane & 15)][0], ks, lane),
                        b_frag_bf(&Kt[nt * 16 + (lane & 15)][0], ks, lane), acc);
#pragma unroll
      for (int r = 0; r < 8; ++r) {
        int t = m0 + r + ((lane & 16) ? 8 : 0);
        int s = nt * 16 + (lane & 15);
        Pm[t][s] = f2bf((s <= t) ? acc[r] : 0.f);
      }
    }
    __syncthreads();

    // --- GEMM2: Y = P @ V + Qs @ S^T  (direct bf16 store to global) ---
#pragma unroll
    for (int nt = 0; nt < 4; ++nt) {
      v8f acc = (v8f){0, 0, 0, 0, 0, 0, 0, 0};
#pragma unroll
      for (int ks = 0; ks < 64; ks += 32)
        acc = wmma_bf16(a_frag_bf(&Pm[m0 + (lane & 15)][0], ks, lane),
                        b_frag_bf(&Vt[nt * 16 + (lane & 15)][0], ks, lane), acc);
#pragma unroll
      for (int ks = 0; ks < 64; ks += 32)
        acc = wmma_bf16(a_frag_bf(&Qs[m0 + (lane & 15)][0], ks, lane),
                        b_frag_f32(&Sm[nt * 16 + (lane & 15)][0], ks, lane), acc);
#pragma unroll
      for (int r = 0; r < 8; ++r) {
        int t = m0 + r + ((lane & 16) ? 8 : 0);
        int i = nt * 16 + (lane & 15);
        Yb[(size_t)(b * T_ + t0 + t) * INNER_ + h * 64 + i] = f2bf(acc[r]);
      }
    }
    __syncthreads();

    // --- GEMM3: S = A_L * (S + V^T @ Kt) ---
    const float aL = __expf(la[63]);
#pragma unroll
    for (int nt = 0; nt < 4; ++nt) {
      v8f acc = (v8f){0, 0, 0, 0, 0, 0, 0, 0};
#pragma unroll
      for (int ks = 0; ks < 64; ks += 32)
        acc = wmma_bf16(a_frag_bf(&Vt[m0 + (lane & 15)][0], ks, lane),
                        b_frag_strided(&Kt[0][nt * 16 + (lane & 15)], 64, ks, lane),
                        acc);
#pragma unroll
      for (int r = 0; r < 8; ++r) {
        int i = m0 + r + ((lane & 16) ? 8 : 0);
        int j = nt * 16 + (lane & 15);
        Sm[i][j] = aL * (Sm[i][j] + acc[r]);
      }
    }
    __syncthreads();
  }
  for (int idx = tid; idx < 4096; idx += 128)
    Sout[(size_t)((b * H_ + h) * 64 + (idx >> 6)) * 64 + (idx & 63)] =
        Sm[idx >> 6][idx & 63];
}

// ---------------- launcher ----------------
extern "C" void kernel_launch(void* const* d_in, const int* in_sizes, int n_in,
                              void* d_out, int out_size, void* d_ws, size_t ws_size,
                              hipStream_t stream) {
  (void)in_sizes; (void)n_in; (void)out_size; (void)ws_size;
  const float* x  = (const float*)d_in[0];
  const float* S0 = (const float*)d_in[1];
  const float* Wq = (const float*)d_in[2];
  const float* Wk = (const float*)d_in[3];
  const float* Wv = (const float*)d_in[4];
  const float* Wa = (const float*)d_in[5];
  const float* ba = (const float*)d_in[6];
  const float* Wb = (const float*)d_in[7];
  const float* bb = (const float*)d_in[8];
  const float* Wo = (const float*)d_in[9];

  char* ws = (char*)d_ws;
  size_t off = 0;
  auto alloc = [&](size_t bytes) -> void* {
    void* p = ws + off;
    off = (off + bytes + 255) & ~(size_t)255;
    return p;
  };
  __bf16* xb   = (__bf16*)alloc((size_t)ROWS_ * DM_ * 2);
  __bf16* Wqb  = (__bf16*)alloc((size_t)DM_ * INNER_ * 2);
  __bf16* Wkb  = (__bf16*)alloc((size_t)DM_ * INNER_ * 2);
  __bf16* Wvb  = (__bf16*)alloc((size_t)DM_ * INNER_ * 2);
  __bf16* Wob  = (__bf16*)alloc((size_t)INNER_ * DM_ * 2);
  __bf16* Qb   = (__bf16*)alloc((size_t)ROWS_ * INNER_ * 2);
  __bf16* Kb   = (__bf16*)alloc((size_t)ROWS_ * INNER_ * 2);
  __bf16* Vb   = (__bf16*)alloc((size_t)ROWS_ * INNER_ * 2);
  __bf16* Yb   = (__bf16*)alloc((size_t)ROWS_ * INNER_ * 2);
  float*  alp  = (float*)alloc((size_t)ROWS_ * H_ * 4);
  float*  bet  = (float*)alloc((size_t)ROWS_ * H_ * 4);

  float* outp  = (float*)d_out;                              // (B,T,DM)
  float* Sout  = (float*)d_out + (size_t)ROWS_ * DM_;        // (B,H,D,D)

  // 1) bf16 conversions
  f32_to_bf16_kernel<<<1024, 256, 0, stream>>>(x,  xb,  (size_t)ROWS_ * DM_);
  f32_to_bf16_kernel<<<256,  256, 0, stream>>>(Wq, Wqb, (size_t)DM_ * INNER_);
  f32_to_bf16_kernel<<<256,  256, 0, stream>>>(Wk, Wkb, (size_t)DM_ * INNER_);
  f32_to_bf16_kernel<<<256,  256, 0, stream>>>(Wv, Wvb, (size_t)DM_ * INNER_);
  f32_to_bf16_kernel<<<256,  256, 0, stream>>>(Wo, Wob, (size_t)INNER_ * DM_);

  // 2) gates
  gates_kernel<<<(ROWS_ * H_ + 255) / 256, 256, 0, stream>>>(x, Wa, ba, Wb, bb, alp, bet);

  // 3) QKV projections (WMMA, bf16 out)
  dim3 ggrid(INNER_ / 64, ROWS_ / 128);
  gemm_bf16_kernel<true><<<ggrid, 256, 0, stream>>>(xb, Wqb, Qb, ROWS_, INNER_, DM_);
  gemm_bf16_kernel<true><<<ggrid, 256, 0, stream>>>(xb, Wkb, Kb, ROWS_, INNER_, DM_);
  gemm_bf16_kernel<true><<<ggrid, 256, 0, stream>>>(xb, Wvb, Vb, ROWS_, INNER_, DM_);

  // 4) chunked recurrent scan (WMMA + TDM staging), writes Yb + final state
  scan_kernel<<<B_ * H_, 128, 0, stream>>>(Qb, Kb, Vb, alp, bet, S0, Yb, Sout);

  // 5) output projection (WMMA, fp32 out straight to d_out)
  dim3 ogrid(DM_ / 64, ROWS_ / 128);
  gemm_bf16_kernel<false><<<ogrid, 256, 0, stream>>>(Yb, Wob, outp, ROWS_, DM_, INNER_);
}
